// NeuralNetworkSimplified_5368709120126
// MI455X (gfx1250) — 8-bit matrix operand order repaired and checked by reading, compiled, not run on hardware
//
#include <hip/hip_runtime.h>
#include <hip/hip_bf16.h>
#include <stdint.h>


typedef __attribute__((ext_vector_type(16))) int      v16i;
typedef __attribute__((ext_vector_type(8)))  float    v8f;
typedef __attribute__((ext_vector_type(16))) __bf16   v16bf;
typedef __attribute__((ext_vector_type(8)))  __bf16   v8bf;
typedef __attribute__((ext_vector_type(4)))  unsigned v4u;
typedef __attribute__((ext_vector_type(8)))  int      v8i_;
typedef __attribute__((ext_vector_type(4)))  int      v4i;

#define BATCH 8192
#define DIM   4096
#define NCLS  1000
#define NPAD  1008
#define EPSV  1e-5f

#define FP8_P1 ((uint8_t)0x38)
#define FP8_M1 ((uint8_t)0xB8)

#define A_PITCH 136
#define B_PITCH 144


__global__ void k_pack_sign(const float* __restrict__ src,
                            uint8_t* __restrict__ dst, int n4) {
  int i = blockIdx.x * blockDim.x + threadIdx.x;
  const int stride = gridDim.x * blockDim.x;
  for (; i < n4; i += stride) {
    const float4 v = ((const float4*)src)[i];
    uchar4 o;
    o.x = (v.x >= 0.f) ? FP8_P1 : FP8_M1;
    o.y = (v.y >= 0.f) ? FP8_P1 : FP8_M1;
    o.z = (v.z >= 0.f) ? FP8_P1 : FP8_M1;
    o.w = (v.w >= 0.f) ? FP8_P1 : FP8_M1;
    ((uchar4*)dst)[i] = o;
  }
}

__global__ void k_bn_sign_pack(const float* __restrict__ Y,
                               const float* __restrict__ a,
                               const float* __restrict__ c,
                               uint8_t* __restrict__ dst, int n4, int N) {
  int i = blockIdx.x * blockDim.x + threadIdx.x;
  const int stride = gridDim.x * blockDim.x;
  for (; i < n4; i += stride) {
    const float4 v  = ((const float4*)Y)[i];
    const int    nc = ((i << 2) % N) >> 2;
    const float4 av = ((const float4*)a)[nc];
    const float4 cv = ((const float4*)c)[nc];
    uchar4 o;
    o.x = (av.x * v.x + cv.x >= 0.f) ? FP8_P1 : FP8_M1;
    o.y = (av.y * v.y + cv.y >= 0.f) ? FP8_P1 : FP8_M1;
    o.z = (av.z * v.z + cv.z >= 0.f) ? FP8_P1 : FP8_M1;
    o.w = (av.w * v.w + cv.w >= 0.f) ? FP8_P1 : FP8_M1;
    ((uchar4*)dst)[i] = o;
  }
}


__global__ void k_col_stats(const float* __restrict__ Y,
                            const float* __restrict__ g,
                            const float* __restrict__ be,
                            float* __restrict__ a, float* __restrict__ c,
                            int M, int N) {
  const int nl = threadIdx.x & 63;
  const int rg = threadIdx.x >> 6;
  const int n  = blockIdx.x * 64 + nl;
  float s = 0.f, s2 = 0.f;
  for (int r = rg; r < M; r += 4) {
    const float v = Y[(size_t)r * N + n];
    s += v; s2 += v * v;
  }
  __shared__ float sh[8][64];
  sh[rg][nl] = s; sh[rg + 4][nl] = s2;
  __syncthreads();
  if (rg == 0) {
    s  = sh[0][nl] + sh[1][nl] + sh[2][nl] + sh[3][nl];
    s2 = sh[4][nl] + sh[5][nl] + sh[6][nl] + sh[7][nl];
    const float mu  = s / (float)M;
    const float var = s2 / (float)M - mu * mu;
    const float rs  = rsqrtf(var + EPSV);
    const float an  = g[n] * rs;
    a[n] = an;
    c[n] = be[n] - an * mu;
  }
}


__device__ __forceinline__ void tdm_load_2d(uint32_t lds_addr,
                                            const uint8_t* gptr,
                                            uint32_t tile_d0,
                                            uint32_t tile_d1,
                                            uint32_t tensor_d0,
                                            uint32_t tensor_d1,
                                            uint32_t stride0,
                                            uint32_t pad_int_code,
                                            uint32_t pad_amt_code) {
  const uint64_t ga = (uint64_t)(uintptr_t)gptr;
  v4u g0;
  g0[0] = 1u;
  g0[1] = lds_addr;
  g0[2] = (uint32_t)(ga & 0xFFFFFFFFu);
  g0[3] = (uint32_t)((ga >> 32) & 0x01FFFFFFu) | (2u << 30);
  v8i_ g1;
  g1[0] = (int)((1u << 20) | (pad_int_code << 22) | (pad_amt_code << 25));
  g1[1] = (int)((tensor_d0 & 0xFFFFu) << 16);
  g1[2] = (int)(((tensor_d0 >> 16) & 0xFFFFu) | ((tensor_d1 & 0xFFFFu) << 16));
  g1[3] = (int)(((tensor_d1 >> 16) & 0xFFFFu) | ((tile_d0 & 0xFFFFu) << 16));
  g1[4] = (int)(tile_d1 & 0xFFFFu);
  g1[5] = (int)stride0;
  g1[6] = (int)((stride0 & 0xFFFFu) << 16);
  g1[7] = (int)(stride0 >> 16);
#pragma unroll
  for (int i = 0; i < 4; ++i) g0[i] = (unsigned)__builtin_amdgcn_readfirstlane((int)g0[i]);
#pragma unroll
  for (int i = 0; i < 8; ++i) g1[i] = __builtin_amdgcn_readfirstlane(g1[i]);
  const v4i z4 = {0, 0, 0, 0};
  const v8i_ z8 = {0, 0, 0, 0, 0, 0, 0, 0};
  __builtin_amdgcn_tensor_load_to_lds(g0, g1, z4, z4, z8, 0);
}


__global__ __launch_bounds__(256)
void k_gemm_fp8(const uint8_t* __restrict__ A, const uint8_t* __restrict__ W,
                const float* __restrict__ bias, float* __restrict__ Y,
                int M, int N, int K) {
  __shared__ uint8_t shA[2][64 * A_PITCH];
  __shared__ uint8_t shB[2][128 * B_PITCH];

  const int wave = threadIdx.x >> 5;
  const int lane = threadIdx.x & 31;
  const int ntiles = N >> 7;
  const int tm = blockIdx.x / ntiles;
  const int tn = blockIdx.x % ntiles;
  const int m0 = tm << 6;
  const int n0 = tn << 7;
  const int ml = lane & 15;
  const int hf = lane >> 4;

  const uint8_t* gA = A + (size_t)m0 * K;
  const uint8_t* gB = W + (size_t)n0 * K;

  if (wave == 0) {
    tdm_load_2d((uint32_t)(uintptr_t)&shA[0][0], gA, 128, 64,
                (uint32_t)K, (uint32_t)M, (uint32_t)K, 4u, 1u);
    tdm_load_2d((uint32_t)(uintptr_t)&shB[0][0], gB, 128, 128,
                (uint32_t)K, (uint32_t)N, (uint32_t)K, 4u, 3u);
  }

  v8f acc[4] = {};
  int cur = 0;
  for (int k = 0; k < K; k += 128) {
    if (wave == 0) __builtin_amdgcn_s_wait_tensorcnt(0);
    __syncthreads();
    if (wave == 0 && k + 128 < K) {
      tdm_load_2d((uint32_t)(uintptr_t)&shA[cur ^ 1][0], gA + k + 128, 128, 64,
                  (uint32_t)K, (uint32_t)M, (uint32_t)K, 4u, 1u);
      tdm_load_2d((uint32_t)(uintptr_t)&shB[cur ^ 1][0], gB + k + 128, 128, 128,
                  (uint32_t)K, (uint32_t)N, (uint32_t)K, 4u, 3u);
    }

    const uint8_t* bp = &shB[cur][(wave * 16 + ml) * B_PITCH + hf * 16];
    v16i bf;
#pragma unroll
    for (int j = 0; j < 4; ++j) {
      const uint4 t = *(const uint4*)(bp + j * 32);
      bf[4 * j + 0] = (int)t.x; bf[4 * j + 1] = (int)t.y;
      bf[4 * j + 2] = (int)t.z; bf[4 * j + 3] = (int)t.w;
    }
#pragma unroll
    for (int mt = 0; mt < 4; ++mt) {
      const uint8_t* ap = &shA[cur][(mt * 16 + ml) * A_PITCH + hf * 16];
      v16i af;
#pragma unroll
      for (int i = 0; i < 8; ++i) {
        const int2 t = *(const int2*)(ap + (i >> 1) * 32 + (i & 1) * 8);
        af[2 * i] = t.x; af[2 * i + 1] = t.y;
      }
      acc[mt] = __builtin_amdgcn_wmma_f32_16x16x128_fp8_fp8(
          af, bf, (short)0, acc[mt], false, false);
    }
    cur ^= 1;
  }

  const int nc = n0 + wave * 16 + ml;
  const float bs = (bias[nc] >= 0.f) ? 1.f : -1.f;
#pragma unroll
  for (int mt = 0; mt < 4; ++mt)
#pragma unroll
    for (int i = 0; i < 8; ++i) {
      const int row = m0 + mt * 16 + i + hf * 8;
      Y[(size_t)row * N + nc] = acc[mt][i] + bs;
    }
}


__device__ __forceinline__ uint16_t f2bf_rne(float f) {
  uint32_t u = __float_as_uint(f);
  u += 0x7FFFu + ((u >> 16) & 1u);
  return (uint16_t)(u >> 16);
}
__device__ __forceinline__ float bf2f(uint16_t h) {
  return __uint_as_float(((uint32_t)h) << 16);
}

__global__ void k_bn_clip_pack_bf16(const float* __restrict__ Y,
                                    const float* __restrict__ a,
                                    const float* __restrict__ c,
                                    uint16_t* __restrict__ Ahi,
                                    uint16_t* __restrict__ Alo,
                                    int n4, int N) {
  int i = blockIdx.x * blockDim.x + threadIdx.x;
  const int stride = gridDim.x * blockDim.x;
  for (; i < n4; i += stride) {
    const float4 v  = ((const float4*)Y)[i];
    const int    nc = ((i << 2) % N) >> 2;
    const float4 av = ((const float4*)a)[nc];
    const float4 cv = ((const float4*)c)[nc];
    float z[4] = { av.x * v.x + cv.x, av.y * v.y + cv.y,
                   av.z * v.z + cv.z, av.w * v.w + cv.w };
    ushort4 oh, ol;
    uint16_t* ph = &oh.x;  uint16_t* pl = &ol.x;
#pragma unroll
    for (int j = 0; j < 4; ++j) {
      const float zc = fminf(1.f, fmaxf(-1.f, z[j]));
      const uint16_t h = f2bf_rne(zc);
      ph[j] = h;
      pl[j] = f2bf_rne(zc - bf2f(h));
    }
    ((ushort4*)Ahi)[i] = oh;
    ((ushort4*)Alo)[i] = ol;
  }
}

__global__ void k_pack_w4_bf16(const float* __restrict__ W4,
                               uint16_t* __restrict__ Whi,
                               uint16_t* __restrict__ Wlo, int n4, int K) {
  int i = blockIdx.x * blockDim.x + threadIdx.x;
  const int stride = gridDim.x * blockDim.x;
  for (; i < n4; i += stride) {
    const int row = (i << 2) / K;
    ushort4 oh = {0, 0, 0, 0}, ol = {0, 0, 0, 0};
    if (row < NCLS) {
      const float4 v = ((const float4*)W4)[i];
      const float  z[4] = { v.x, v.y, v.z, v.w };
      uint16_t* ph = &oh.x;  uint16_t* pl = &ol.x;
#pragma unroll
      for (int j = 0; j < 4; ++j) {
        const uint16_t h = f2bf_rne(z[j]);
        ph[j] = h;
        pl[j] = f2bf_rne(z[j] - bf2f(h));
      }
    }
    ((ushort4*)Whi)[i] = oh;
    ((ushort4*)Wlo)[i] = ol;
  }
}


__device__ __forceinline__ v16bf load_frag16(const uint16_t* p) {
  const uint4 c0 = *(const uint4*)(p);
  const uint4 c1 = *(const uint4*)(p + 16);
  const v8bf l0 = __builtin_bit_cast(v8bf, c0);
  const v8bf l1 = __builtin_bit_cast(v8bf, c1);
  v16bf r;
#pragma unroll
  for (int i = 0; i < 8; ++i) { r[i] = l0[i]; r[i + 8] = l1[i]; }
  return r;
}

__global__ __launch_bounds__(256)
void k_gemm_bf16_final(const uint16_t* __restrict__ Ahi,
                       const uint16_t* __restrict__ Alo,
                       const uint16_t* __restrict__ Whi,
                       const uint16_t* __restrict__ Wlo,
                       const float* __restrict__ b4,
                       float* __restrict__ out, int M, int K) {
  const int wave = threadIdx.x >> 5;
  const int lane = threadIdx.x & 31;
  const int gid  = blockIdx.x * 8 + wave;
  const int ntiles = NPAD >> 4;
  const int tm = gid / ntiles;
  const int tn = gid % ntiles;
  const int m0 = tm << 6;
  const int n0 = tn << 4;
  const int ml = lane & 15;
  const int hf = lane >> 4;

  const uint16_t* ah = Ahi + (size_t)(m0 + ml) * K + hf * 8;
  const uint16_t* al = Alo + (size_t)(m0 + ml) * K + hf * 8;
  const uint16_t* wh = Whi + (size_t)(n0 + ml) * K + hf * 8;
  const uint16_t* wl = Wlo + (size_t)(n0 + ml) * K + hf * 8;

  v8f acc[4] = {};
  for (int k = 0; k < K; k += 32) {
    const v16bf bh = load_frag16(wh + k);
    const v16bf bl = load_frag16(wl + k);
#pragma unroll
    for (int mt = 0; mt < 4; ++mt) {
      const size_t ro = (size_t)mt * 16 * K + k;
      const v16bf xh = load_frag16(ah + ro);
      const v16bf xl = load_frag16(al + ro);
      acc[mt] = __builtin_amdgcn_wmma_f32_16x16x32_bf16(
          false, xh, false, bh, (short)0, acc[mt], false, false);
      acc[mt] = __builtin_amdgcn_wmma_f32_16x16x32_bf16(
          false, xh, false, bl, (short)0, acc[mt], false, false);
      acc[mt] = __builtin_amdgcn_wmma_f32_16x16x32_bf16(
          false, xl, false, bh, (short)0, acc[mt], false, false);
    }
  }

  const int n = n0 + ml;
  if (n < NCLS) {
    const float bb = b4[n];
#pragma unroll
    for (int mt = 0; mt < 4; ++mt)
#pragma unroll
      for (int i = 0; i < 8; ++i) {
        const int row = m0 + mt * 16 + i + hf * 8;
        out[(size_t)row * NCLS + n] = acc[mt][i] + bb;
      }
  }
}


extern "C" void kernel_launch(void* const* d_in, const int* in_sizes, int n_in,
                              void* d_out, int out_size, void* d_ws, size_t ws_size,
                              hipStream_t stream) {
  const float* x   = (const float*)d_in[0];
  const float* W1  = (const float*)d_in[2];
  const float* b1  = (const float*)d_in[3];
  const float* g1  = (const float*)d_in[4];
  const float* be1 = (const float*)d_in[5];
  const float* W2  = (const float*)d_in[6];
  const float* b2  = (const float*)d_in[7];
  const float* g2  = (const float*)d_in[8];
  const float* be2 = (const float*)d_in[9];
  const float* W3  = (const float*)d_in[10];
  const float* b3  = (const float*)d_in[11];
  const float* g3  = (const float*)d_in[12];
  const float* be3 = (const float*)d_in[13];
  const float* W4  = (const float*)d_in[14];
  const float* b4  = (const float*)d_in[15];
  float* out = (float*)d_out;

  uint8_t* ws = (uint8_t*)d_ws;
  size_t off = 0;
  auto alloc = [&](size_t bytes) {
    uint8_t* p = ws + off;
    off += (bytes + 255) & ~(size_t)255;
    return p;
  };
  const size_t WSZ = (size_t)DIM * DIM;
  uint8_t*  wpk1 = alloc(WSZ);
  uint8_t*  wpk2 = alloc(WSZ);
  uint8_t*  wpk3 = alloc(WSZ);
  uint8_t*  apk  = alloc((size_t)BATCH * DIM);
  float*    Ybuf = (float*)alloc((size_t)BATCH * DIM * 4);
  float*    acoef = (float*)alloc(DIM * 4);
  float*    ccoef = (float*)alloc(DIM * 4);
  uint16_t* Ahi  = (uint16_t*)alloc((size_t)BATCH * DIM * 2);
  uint16_t* Alo  = (uint16_t*)alloc((size_t)BATCH * DIM * 2);
  uint16_t* Whi  = (uint16_t*)alloc((size_t)NPAD * DIM * 2);
  uint16_t* Wlo  = (uint16_t*)alloc((size_t)NPAD * DIM * 2);
  (void)ws_size; (void)in_sizes; (void)n_in; (void)out_size;

  const int TPB = 256;
  const int w4cnt   = (int)(WSZ / 4);
  const int act4cnt = (int)((size_t)BATCH * DIM / 4);
  const int w44cnt  = (int)((size_t)NPAD * DIM / 4);
  const int gemmBlocks  = (BATCH / 64) * (DIM / 128);
  const int gemmFBlocks = (BATCH / 64) * (NPAD / 16) / 8;

  k_pack_sign<<<2048, TPB, 0, stream>>>(W1, wpk1, w4cnt);
  k_pack_sign<<<2048, TPB, 0, stream>>>(W2, wpk2, w4cnt);
  k_pack_sign<<<2048, TPB, 0, stream>>>(W3, wpk3, w4cnt);
  k_pack_sign<<<2048, TPB, 0, stream>>>(x, apk, act4cnt);

  k_gemm_fp8<<<gemmBlocks, TPB, 0, stream>>>(apk, wpk1, b1, Ybuf, BATCH, DIM, DIM);
  k_col_stats<<<DIM / 64, TPB, 0, stream>>>(Ybuf, g1, be1, acoef, ccoef, BATCH, DIM);
  k_bn_sign_pack<<<2048, TPB, 0, stream>>>(Ybuf, acoef, ccoef, apk, act4cnt, DIM);

  k_gemm_fp8<<<gemmBlocks, TPB, 0, stream>>>(apk, wpk2, b2, Ybuf, BATCH, DIM, DIM);
  k_col_stats<<<DIM / 64, TPB, 0, stream>>>(Ybuf, g2, be2, acoef, ccoef, BATCH, DIM);
  k_bn_sign_pack<<<2048, TPB, 0, stream>>>(Ybuf, acoef, ccoef, apk, act4cnt, DIM);

  k_gemm_fp8<<<gemmBlocks, TPB, 0, stream>>>(apk, wpk3, b3, Ybuf, BATCH, DIM, DIM);
  k_col_stats<<<DIM / 64, TPB, 0, stream>>>(Ybuf, g3, be3, acoef, ccoef, BATCH, DIM);
  k_bn_clip_pack_bf16<<<2048, TPB, 0, stream>>>(Ybuf, acoef, ccoef, Ahi, Alo,
                                                act4cnt, DIM);

  k_pack_w4_bf16<<<2048, TPB, 0, stream>>>(W4, Whi, Wlo, w44cnt, DIM);
  k_gemm_bf16_final<<<gemmFBlocks, TPB, 0, stream>>>(Ahi, Alo, Whi, Wlo, b4,
                                                     out, BATCH, DIM);
}
